// TransformerBlock_23141283791312
// MI455X (gfx1250) — matrix operand order repaired and checked by reading, not compiled, not run on hardware
//
#include <hip/hip_runtime.h>
#include <math.h>

typedef _Float16 h16;
typedef __attribute__((ext_vector_type(4)))  _Float16 v4h;
typedef __attribute__((ext_vector_type(8)))  _Float16 v8h;
typedef __attribute__((ext_vector_type(16))) _Float16 v16h;
typedef __attribute__((ext_vector_type(8)))  float    v8f;
typedef int v4i32 __attribute__((vector_size(4 * sizeof(int))));

#define EMBED    768
#define HEADS    12
#define HEAD_DIM 64
#define HIDDEN   3072
#define BATCH    8
#define SEQ      1024
#define MROWS    (BATCH*SEQ)

#if defined(__AMDGCN__) && __has_builtin(__builtin_amdgcn_global_load_async_to_lds_b128) && __has_builtin(__builtin_amdgcn_s_wait_asynccnt)
#define USE_ASYNC_LDS 1
#else
#define USE_ASYNC_LDS 0
#endif

static __device__ __forceinline__ void async_b128(const h16* g, h16* l){
#if USE_ASYNC_LDS
  __builtin_amdgcn_global_load_async_to_lds_b128(
      (__attribute__((address_space(1))) v4i32*)(g),
      (__attribute__((address_space(3))) v4i32*)(l), 0, 0);
#else
  *(uint4*)l = *(const uint4*)g;
#endif
}
static __device__ __forceinline__ void wait_async(){
#if USE_ASYNC_LDS
  __builtin_amdgcn_s_wait_asynccnt(0);
#endif
}

static __device__ __forceinline__ v16h combine16(v8h lo, v8h hi){
  v16h r;
#pragma unroll
  for (int i = 0; i < 8; ++i){ r[i] = lo[i]; r[8+i] = hi[i]; }
  return r;
}
static __device__ __forceinline__ float redmax16(float v){
#pragma unroll
  for (int m = 1; m < 16; m <<= 1) v = fmaxf(v, __shfl_xor(v, m, 32));
  return v;
}
static __device__ __forceinline__ float redsum16(float v){
#pragma unroll
  for (int m = 1; m < 16; m <<= 1) v += __shfl_xor(v, m, 32);
  return v;
}
static __device__ __forceinline__ float redsum32(float v){
#pragma unroll
  for (int m = 1; m < 32; m <<= 1) v += __shfl_xor(v, m, 32);
  return v;
}

// ---------------------------------------------------------------- f32 -> f16
__global__ __launch_bounds__(256) void cvt_f32_f16_kernel(
    const float* __restrict__ src, h16* __restrict__ dst, int n4){
  int i = blockIdx.x * 256 + threadIdx.x;
  if (i < n4){
    float4 v = ((const float4*)src)[i];
    v4h r = { (h16)v.x, (h16)v.y, (h16)v.z, (h16)v.w };
    ((v4h*)dst)[i] = r;
  }
}

// ------------------------------------------------- LayerNorm, one wave / row
__global__ __launch_bounds__(256) void ln_f16_kernel(
    const float* __restrict__ x, const float* __restrict__ g,
    const float* __restrict__ b, h16* __restrict__ out){
  int wave = (blockIdx.x * 256 + threadIdx.x) >> 5;
  int lane = threadIdx.x & 31;
  const float* row = x + (size_t)wave * EMBED;
  float s = 0.f, ss = 0.f;
  for (int i = lane; i < EMBED; i += 32){ float v = row[i]; s += v; ss += v*v; }
  s  = redsum32(s);
  ss = redsum32(ss);
  float mu  = s  * (1.0f/EMBED);
  float var = ss * (1.0f/EMBED) - mu*mu;
  float rs  = rsqrtf(var + 1e-5f);
  h16* orow = out + (size_t)wave * EMBED;
  for (int i = lane; i < EMBED; i += 32)
    orow[i] = (h16)((row[i] - mu) * rs * g[i] + b[i]);
}

// ---------------------------------------------------------------- GEMM (NT)
// C[M,N] = act(A[M,K] * W[N,K]^T + bias) (+ residual)
// Block tile 128(M) x 256(N), BK=32, double-buffered async-to-LDS staging.
// 8 waves, each wave 64x64 -> 16 WMMAs per K-step.
// flags: bit0 = GELU, bit1 = f32 output (else f16)
__global__ __launch_bounds__(256) void gemm_f16_kernel(
    const h16* __restrict__ A, const h16* __restrict__ W,
    const float* __restrict__ bias, void* __restrict__ Cout,
    const float* __restrict__ residual, int M, int N, int K, int flags){
  __shared__ h16 As[2][128][40];   // 32 K + 8 pad -> 80B rows (16B aligned)
  __shared__ h16 Bs[2][256][40];
  const int t    = threadIdx.x;
  const int lane = t & 31, lsub = lane & 15, hi = lane >> 4;
  const int wave = t >> 5, waveM = wave >> 2, waveN = wave & 3;
  const int m0 = blockIdx.y * 128, n0 = blockIdx.x * 256;
  const int rowA = t >> 1, colA = (t & 1) * 16;

  const h16* gA = A + (size_t)(m0 + rowA) * K + colA;
  const h16* gB = W + (size_t)(n0 + t) * K;

  v8f zero = {};
  v8f acc[4][4];
#pragma unroll
  for (int i = 0; i < 4; ++i)
#pragma unroll
    for (int j = 0; j < 4; ++j) acc[i][j] = zero;

  // stage K-slab k0 into buffer `buf`
  auto stage = [&](int k0, int buf){
    async_b128(gA + k0,      &As[buf][rowA][colA]);
    async_b128(gA + k0 + 8,  &As[buf][rowA][colA + 8]);
    async_b128(gB + k0,      &Bs[buf][t][0]);
    async_b128(gB + k0 + 8,  &Bs[buf][t][8]);
    async_b128(gB + k0 + 16, &Bs[buf][t][16]);
    async_b128(gB + k0 + 24, &Bs[buf][t][24]);
  };

  stage(0, 0);
  int buf = 0;
  for (int k0 = 0; k0 < K; k0 += 32){
    wait_async();        // our copies for `buf` are in LDS
    __syncthreads();     // everyone's copies visible; prev compute finished
    if (k0 + 32 < K) stage(k0 + 32, buf ^ 1);   // overlap next copy w/ compute

    v16h bfr[4];
#pragma unroll
    for (int nt = 0; nt < 4; ++nt){
      int bn = waveN*64 + nt*16 + lsub;
      v8h lo = *(const v8h*)&Bs[buf][bn][hi*8];
      v8h hh = *(const v8h*)&Bs[buf][bn][16 + hi*8];
      bfr[nt] = combine16(lo, hh);
    }
#pragma unroll
    for (int mt = 0; mt < 4; ++mt){
      int am = waveM*64 + mt*16 + lsub;
      v8h lo = *(const v8h*)&As[buf][am][hi*8];
      v8h hh = *(const v8h*)&As[buf][am][16 + hi*8];
      v16h afr = combine16(lo, hh);
#pragma unroll
      for (int nt = 0; nt < 4; ++nt){
        acc[mt][nt] = __builtin_amdgcn_wmma_f32_16x16x32_f16(
            false, afr, false, bfr[nt], (short)0, acc[mt][nt], false, false);
      }
    }
    buf ^= 1;
  }

  // epilogue: bias (+gelu) (+residual), f16 or f32 store
#pragma unroll
  for (int mt = 0; mt < 4; ++mt){
    int grow0 = m0 + waveM*64 + mt*16 + hi*8;
#pragma unroll
    for (int nt = 0; nt < 4; ++nt){
      int gcol = n0 + waveN*64 + nt*16 + lsub;
      float bv = bias[gcol];
      v8f a = acc[mt][nt];
#pragma unroll
      for (int r = 0; r < 8; ++r){
        int grow = grow0 + r;
        float v = a[r] + bv;
        if (flags & 1) v = 0.5f * v * (1.0f + erff(v * 0.70710678118f));
        if (residual)  v += residual[(size_t)grow * N + gcol];
        if (flags & 2) ((float*)Cout)[(size_t)grow * N + gcol] = v;
        else           ((h16*)Cout)[(size_t)grow * N + gcol] = (h16)v;
      }
    }
  }
}

// --------------------------------------------------- flash attention (f16)
// qkv: [B*SEQ, 3*EMBED] f16, col = s*768 + h*64 + d ; ctx: [B*SEQ, EMBED]
__global__ __launch_bounds__(256) void attn_f16_kernel(
    const h16* __restrict__ qkv, h16* __restrict__ ctx){
  __shared__ h16 Ks[32][72];      // 32 keys x 64 hd (+8 pad)
  __shared__ h16 Vt[64][40];      // 64 hd x 32 keys (+8 pad), transposed V
  __shared__ h16 Ps[8][16][40];   // per-wave P re-layout scratch

  const int t    = threadIdx.x;
  const int lane = t & 31, lsub = lane & 15, hi = lane >> 4;
  const int w    = t >> 5;
  const int bh = blockIdx.x, bb = bh / HEADS, hh = bh % HEADS;
  const int q0 = blockIdx.y * 128;
  const size_t rowBase = (size_t)bb * SEQ;
  const int qOff = hh * HEAD_DIM;
  const int kOff = EMBED + hh * HEAD_DIM;
  const int vOff = 2*EMBED + hh * HEAD_DIM;

  // Q tile for this wave (16 queries x 64 hd) as two A fragments
  v16h aq[2];
  {
    const h16* pq = qkv + (rowBase + q0 + w*16 + lsub) * (size_t)(3*EMBED) + qOff;
#pragma unroll
    for (int j = 0; j < 2; ++j){
      v8h lo = *(const v8h*)(pq + j*32 + hi*8);
      v8h hh2 = *(const v8h*)(pq + j*32 + 16 + hi*8);
      aq[j] = combine16(lo, hh2);
    }
  }

  v8f zero = {};
  v8f o[4];
#pragma unroll
  for (int i = 0; i < 4; ++i) o[i] = zero;
  float mrow[8], lrow[8];
#pragma unroll
  for (int r = 0; r < 8; ++r){ mrow[r] = -3.0e38f; lrow[r] = 0.f; }

  const int ldRow = t >> 3;           // 0..31 (key within block)
  const int ldCol = (t & 7) * 8;      // 0..56 (hd chunk)
  const float SC = 0.125f;            // 1/sqrt(64)

  for (int kb = 0; kb < SEQ; kb += 32){
    __syncthreads();                  // prior compute done before restage
    { // stage K block (async direct-to-LDS) and transposed V block
      const h16* gk = qkv + (rowBase + kb + ldRow) * (size_t)(3*EMBED) + kOff + ldCol;
      async_b128(gk, &Ks[ldRow][ldCol]);
      const h16* gv = qkv + (rowBase + kb + ldRow) * (size_t)(3*EMBED) + vOff + ldCol;
      v8h vv = *(const v8h*)gv;
#pragma unroll
      for (int i = 0; i < 8; ++i) Vt[ldCol + i][ldRow] = vv[i];
    }
    wait_async();
    __syncthreads();

    // S = Q K^T  (two 16x16 C tiles covering 32 keys)
    v8f s0 = zero, s1 = zero;
#pragma unroll
    for (int j = 0; j < 2; ++j){
      v8h lo  = *(const v8h*)&Ks[lsub][j*32 + hi*8];
      v8h hh2 = *(const v8h*)&Ks[lsub][j*32 + 16 + hi*8];
      v16h bk = combine16(lo, hh2);
      s0 = __builtin_amdgcn_wmma_f32_16x16x32_f16(false, aq[j], false, bk, (short)0, s0, false, false);
      lo  = *(const v8h*)&Ks[16 + lsub][j*32 + hi*8];
      hh2 = *(const v8h*)&Ks[16 + lsub][j*32 + 16 + hi*8];
      bk = combine16(lo, hh2);
      s1 = __builtin_amdgcn_wmma_f32_16x16x32_f16(false, aq[j], false, bk, (short)0, s1, false, false);
    }

    // online softmax across this 32-key slab
    float alpha[8];
#pragma unroll
    for (int r = 0; r < 8; ++r){
      float v0 = s0[r] * SC, v1 = s1[r] * SC;
      float mx = redmax16(fmaxf(v0, v1));
      float mn = fmaxf(mrow[r], mx);
      float a  = __expf(mrow[r] - mn);
      mrow[r] = mn;
      float p0 = __expf(v0 - mn), p1 = __expf(v1 - mn);
      lrow[r] = lrow[r] * a + redsum16(p0 + p1);
      alpha[r] = a;
      Ps[w][hi*8 + r][lsub]      = (h16)p0;   // C layout -> LDS
      Ps[w][hi*8 + r][16 + lsub] = (h16)p1;
    }
    // reload P as A fragment (16 queries x 32 keys)
    v16h pf;
    {
      v8h lo  = *(const v8h*)&Ps[w][lsub][hi*8];
      v8h hh2 = *(const v8h*)&Ps[w][lsub][16 + hi*8];
      pf = combine16(lo, hh2);
    }
    // O = O*alpha + P @ V  (4 hd tiles of 16)
#pragma unroll
    for (int n = 0; n < 4; ++n){
      v8f a2 = o[n];
#pragma unroll
      for (int r = 0; r < 8; ++r) a2[r] *= alpha[r];
      v8h lo  = *(const v8h*)&Vt[n*16 + lsub][hi*8];
      v8h hh2 = *(const v8h*)&Vt[n*16 + lsub][16 + hi*8];
      v16h bv = combine16(lo, hh2);
      o[n] = __builtin_amdgcn_wmma_f32_16x16x32_f16(false, pf, false, bv, (short)0, a2, false, false);
    }
  }

#pragma unroll
  for (int r = 0; r < 8; ++r) lrow[r] = 1.0f / lrow[r];
#pragma unroll
  for (int n = 0; n < 4; ++n){
#pragma unroll
    for (int r = 0; r < 8; ++r){
      size_t row = rowBase + q0 + w*16 + hi*8 + r;
      ctx[row * EMBED + hh*HEAD_DIM + n*16 + lsub] = (h16)(o[n][r] * lrow[r]);
    }
  }
}

// ----------------------------------------------------------------- launcher
extern "C" void kernel_launch(void* const* d_in, const int* in_sizes, int n_in,
                              void* d_out, int out_size, void* d_ws, size_t ws_size,
                              hipStream_t stream){
  (void)in_sizes; (void)n_in; (void)out_size; (void)ws_size;
  const float* x      = (const float*)d_in[0];
  const float* ln1_g  = (const float*)d_in[1];
  const float* ln1_b  = (const float*)d_in[2];
  const float* qkv_w  = (const float*)d_in[3];
  const float* qkv_b  = (const float*)d_in[4];
  const float* proj_w = (const float*)d_in[5];
  const float* proj_b = (const float*)d_in[6];
  const float* ln2_g  = (const float*)d_in[7];
  const float* ln2_b  = (const float*)d_in[8];
  const float* fc1_w  = (const float*)d_in[9];
  const float* fc1_b  = (const float*)d_in[10];
  const float* fc2_w  = (const float*)d_in[11];
  const float* fc2_b  = (const float*)d_in[12];

  char* ws = (char*)d_ws;
  size_t off = 0;
  auto alloc = [&](size_t bytes) -> void* {
    void* p = ws + off;
    off += (bytes + 255) & ~(size_t)255;
    return p;
  };
  h16*   wqkv  = (h16*)alloc((size_t)3*EMBED*EMBED*sizeof(h16));
  h16*   wproj = (h16*)alloc((size_t)EMBED*EMBED*sizeof(h16));
  h16*   wfc1  = (h16*)alloc((size_t)HIDDEN*EMBED*sizeof(h16));
  h16*   wfc2  = (h16*)alloc((size_t)EMBED*HIDDEN*sizeof(h16));
  h16*   hbuf  = (h16*)alloc((size_t)MROWS*EMBED*sizeof(h16));
  h16*   qkvb  = (h16*)alloc((size_t)MROWS*3*EMBED*sizeof(h16));
  h16*   ctxb  = (h16*)alloc((size_t)MROWS*EMBED*sizeof(h16));
  float* x1    = (float*)alloc((size_t)MROWS*EMBED*sizeof(float));
  h16*   hfc1  = (h16*)alloc((size_t)MROWS*HIDDEN*sizeof(h16));

  auto cvt = [&](const float* s, h16* d, size_t n){
    int n4 = (int)(n / 4);
    cvt_f32_f16_kernel<<<dim3((n4 + 255) / 256), dim3(256), 0, stream>>>(s, d, n4);
  };
  cvt(qkv_w,  wqkv,  (size_t)3*EMBED*EMBED);
  cvt(proj_w, wproj, (size_t)EMBED*EMBED);
  cvt(fc1_w,  wfc1,  (size_t)HIDDEN*EMBED);
  cvt(fc2_w,  wfc2,  (size_t)EMBED*HIDDEN);

  // LN1 -> hbuf (f16)
  ln_f16_kernel<<<dim3(MROWS/8), dim3(256), 0, stream>>>(x, ln1_g, ln1_b, hbuf);
  // QKV GEMM -> qkvb (f16)
  gemm_f16_kernel<<<dim3((3*EMBED)/256, MROWS/128), dim3(256), 0, stream>>>(
      hbuf, wqkv, qkv_b, qkvb, nullptr, MROWS, 3*EMBED, EMBED, 0);
  // attention -> ctxb (f16)
  attn_f16_kernel<<<dim3(BATCH*HEADS, SEQ/128), dim3(256), 0, stream>>>(qkvb, ctxb);
  // proj GEMM + residual(x) -> x1 (f32)
  gemm_f16_kernel<<<dim3(EMBED/256, MROWS/128), dim3(256), 0, stream>>>(
      ctxb, wproj, proj_b, x1, x, MROWS, EMBED, EMBED, 2);
  // LN2 -> hbuf (f16)
  ln_f16_kernel<<<dim3(MROWS/8), dim3(256), 0, stream>>>(x1, ln2_g, ln2_b, hbuf);
  // FC1 GEMM + GELU -> hfc1 (f16)
  gemm_f16_kernel<<<dim3(HIDDEN/256, MROWS/128), dim3(256), 0, stream>>>(
      hbuf, wfc1, fc1_b, hfc1, nullptr, MROWS, HIDDEN, EMBED, 1);
  // FC2 GEMM + residual(x1) -> d_out (f32)
  gemm_f16_kernel<<<dim3(EMBED/256, MROWS/128), dim3(256), 0, stream>>>(
      hfc1, wfc2, fc2_b, d_out, x1, MROWS, EMBED, HIDDEN, 2);
}
